// DCVQVAE_17892833755573
// MI455X (gfx1250) — matrix operand order repaired and checked by reading, not compiled, not run on hardware
//
#include <hip/hip_runtime.h>

typedef __bf16 bf16;
typedef __bf16 v16bf __attribute__((ext_vector_type(16)));
typedef __bf16 v8bf  __attribute__((ext_vector_type(8)));
typedef float  v8f   __attribute__((ext_vector_type(8)));
typedef unsigned int u32x4 __attribute__((ext_vector_type(4)));
typedef int i32x4 __attribute__((ext_vector_type(4)));
typedef int i32x8 __attribute__((ext_vector_type(8)));

#define TBM 128
#define TBN 64
#define TBK 32

// ---------------------------------------------------------------- TDM helper
// Load a [64 rows x 32 cols] bf16 tile of a row-major [rows x cols] matrix
// (row stride = ldK elems) into LDS (compact [64][32]) via the Tensor Data
// Mover.  tensor_dim = remaining extent -> hardware zero-fills OOB reads.
// TDM ignores EXEC; issue from one wave only.
__device__ __forceinline__ void tdm_load_b_tile(const bf16* gbase, int ldK,
                                                int remK, int remN,
                                                unsigned ldsAddr) {
  unsigned long long ga = (unsigned long long)(uintptr_t)gbase;
  u32x4 g0;
  g0[0] = 1u;                                   // count=1 valid descriptor
  g0[1] = ldsAddr;                              // lds_addr (bytes)
  g0[2] = (unsigned)(ga & 0xFFFFFFFFull);       // global_addr[31:0]
  g0[3] = (unsigned)((ga >> 32) & 0x01FFFFFFull) | (2u << 30);  // addr[56:32]|type=2
  i32x8 g1;
  g1[0] = 0x00010000;                           // data_size=1 (2 bytes)
  g1[1] = (int)(((unsigned)remK & 0xFFFFu) << 16);              // tensor_dim0 lo16
  g1[2] = (int)((((unsigned)remK >> 16) & 0xFFFFu) |
                (((unsigned)remN & 0xFFFFu) << 16));            // dim0 hi16 | dim1 lo16
  g1[3] = (int)((((unsigned)remN >> 16) & 0xFFFFu) | (32u << 16)); // dim1 hi | tile_dim0=32
  g1[4] = 64;                                   // tile_dim1=64 rows (tile_dim2=0)
  g1[5] = ldK;                                  // tensor_dim0_stride lo32 (elems)
  g1[6] = 0;
  g1[7] = 0;
  i32x4 gz = {0, 0, 0, 0};
#if defined(__clang_major__) && __clang_major__ >= 23
  i32x8 gz8 = {0, 0, 0, 0, 0, 0, 0, 0};
  __builtin_amdgcn_tensor_load_to_lds(g0, g1, gz, gz, gz8, 0);
#else
  __builtin_amdgcn_tensor_load_to_lds(g0, g1, gz, gz, 0);
#endif
}

// ---------------------------------------------------------------- utilities

__global__ void k_f32_to_bf16(const float* __restrict__ s, bf16* __restrict__ d, int n) {
  int i = blockIdx.x * 256 + threadIdx.x;
  if (i < n) d[i] = (bf16)s[i];
}

__global__ void k_zero_f32(float* __restrict__ p, int n) {
  int i = blockIdx.x * 256 + threadIdx.x;
  if (i < n) p[i] = 0.0f;
}

// Pack ConvTranspose2d weights [IC][OC][4][4] (torch layout) into per-parity
// GEMM-B layout [parity(a*2+b)][OC][IC*4] bf16.
__global__ void k_pack_dect(const float* __restrict__ w, bf16* __restrict__ dst,
                            int IC, int OC) {
  int i = blockIdx.x * 256 + threadIdx.x;
  int K4 = IC * 4;
  int total = 4 * OC * K4;
  if (i >= total) return;
  int p  = i / (OC * K4);
  int r  = i - p * (OC * K4);
  int oc = r / K4;
  int k  = r - oc * K4;
  int ic = k >> 2, t = k & 3, th = t >> 1, tw = t & 1;
  int a = p >> 1, b = p & 1;
  int kh = a ? (th ? 2 : 0) : (th ? 3 : 1);
  int kw = b ? (tw ? 2 : 0) : (tw ? 3 : 1);
  dst[i] = (bf16)w[((long)(ic * OC + oc) * 4 + kh) * 4 + kw];
}

// ---------------------------------------------------------------- conv GEMM
// Software-pipelined: TDM double-buffers the B (weight) tile, A (im2col
// gather) is staged through registers so its global loads overlap compute.

__global__ __launch_bounds__(256) void k_conv_wmma(
    const bf16* __restrict__ in, const bf16* __restrict__ wgt,
    const float* __restrict__ bias, void* __restrict__ out,
    float* __restrict__ statSum, float* __restrict__ statSq,
    int Nb, int Cin, int IH, int IW, int OC, int OH, int OW,
    int KH, int KW, int stride, int pad, int outF32) {
  __shared__ __align__(16) bf16 As[TBM][TBK];
  __shared__ __align__(32) bf16 Bs[2][TBN][TBK];
  __shared__ float cSum[TBN], cSq[TBN];
  __shared__ int tC[TBM];    // n*Cin
  __shared__ int tIHb[TBM];  // oh*stride-pad (poisoned if row invalid)
  __shared__ int tIWb[TBM];
  __shared__ int tNo[TBM];   // n
  __shared__ int tR[TBM];    // oh*OW+ow

  const int tid = threadIdx.x;
  const int M  = Nb * OH * OW;
  const int K  = Cin * KH * KW;
  const int m0 = blockIdx.x * TBM;
  const int n0 = blockIdx.y * TBN;

  if (tid < TBN) { cSum[tid] = 0.0f; cSq[tid] = 0.0f; }
  if (tid < TBM) {            // one-time row decode
    int m = m0 + tid;
    int n = 0, r = 0, ihb = -(1 << 20), iwb = 0;
    if (m < M) {
      n = m / (OH * OW);
      r = m - n * (OH * OW);
      int oh = r / OW, ow = r - (r / OW) * OW;
      ihb = oh * stride - pad;
      iwb = ow * stride - pad;
    }
    tC[tid] = n * Cin; tIHb[tid] = ihb; tIWb[tid] = iwb; tNo[tid] = n; tR[tid] = r;
  }
  __syncthreads();

  const int lane = tid & 31;
  const int wave = tid >> 5;
  const int wm = wave & 3, wn = wave >> 2;
  const int lr = lane & 15, lh = lane >> 4;
  const int kiMine = tid & 31;
  const int miBase = tid >> 5;
  const int KHKW = KH * KW;
  const unsigned ldsB0 = (unsigned)(size_t)&Bs[0][0][0];
  const unsigned ldsB1 = (unsigned)(size_t)&Bs[1][0][0];
  const int nk = (K + TBK - 1) / TBK;

  // gather the 16 A values for K-step kt into registers
  auto loadA = [&](int kt, bf16 (&regs)[16]) {
    int k = kt * TBK + kiMine;
    bool kok = (k < K);
    int c = 0, kh = 0, kw = 0;
    if (kok) {
      c = k / KHKW;
      int kr = k - c * KHKW;
      kh = kr / KW;
      kw = kr - (kr / KW) * KW;
    }
#pragma unroll
    for (int i = 0; i < 16; ++i) {
      int mi = i * 8 + miBase;
      int ih = tIHb[mi] + kh;
      int iw = tIWb[mi] + kw;
      bf16 v = (bf16)0.0f;
      if (kok && (unsigned)ih < (unsigned)IH && (unsigned)iw < (unsigned)IW)
        v = in[((long)(tC[mi] + c) * IH + ih) * IW + iw];
      regs[i] = v;
    }
  };

  v8f acc[2][2] = {};
  bf16 aReg[16];
  loadA(0, aReg);
  if (tid == 0)  // pre-issue first B tile
    tdm_load_b_tile(wgt + (long)n0 * K, K, K, OC - n0, ldsB0);

  for (int kt = 0; kt < nk; ++kt) {
    // drain staged A into LDS (compiler places s_wait_loadcnt here)
#pragma unroll
    for (int i = 0; i < 16; ++i) As[i * 8 + miBase][kiMine] = aReg[i];
    if (tid == 0) {
      if (kt + 1 < nk) {  // stream next B tile behind this step's compute
        tdm_load_b_tile(wgt + (long)n0 * K + (kt + 1) * TBK, K,
                        K - (kt + 1) * TBK, OC - n0, (kt + 1) & 1 ? ldsB1 : ldsB0);
        __builtin_amdgcn_s_wait_tensorcnt(1);  // in-order: tile kt resident
      } else {
        __builtin_amdgcn_s_wait_tensorcnt(0);
      }
    }
    __syncthreads();
    if (kt + 1 < nk) loadA(kt + 1, aReg);  // global loads overlap WMMA below

    v16bf afr[2], bfr[2];
#pragma unroll
    for (int mt = 0; mt < 2; ++mt) {
      int row = wm * 32 + mt * 16 + lr;
      int kh8 = lh * 8;
      v8bf lo = *(const v8bf*)&As[row][2 * kh8];
      v8bf hi = *(const v8bf*)&As[row][2 * kh8 + 8];
#pragma unroll
      for (int j = 0; j < 8; ++j) { afr[mt][j] = lo[j]; afr[mt][8 + j] = hi[j]; }
    }
#pragma unroll
    for (int nt = 0; nt < 2; ++nt) {
      int col = wn * 32 + nt * 16 + lr;
      bfr[nt] = *(const v16bf*)&Bs[kt & 1][col][lh * 16];
    }
#pragma unroll
    for (int mt = 0; mt < 2; ++mt)
#pragma unroll
      for (int nt = 0; nt < 2; ++nt)
        acc[mt][nt] = __builtin_amdgcn_wmma_f32_16x16x32_bf16(
            false, afr[mt], false, bfr[nt], (short)0, acc[mt][nt], false, false);
    __syncthreads();
  }

  // ---- store (+ optional BN stats)
#pragma unroll
  for (int mt = 0; mt < 2; ++mt) {
#pragma unroll
    for (int nt = 0; nt < 2; ++nt) {
      int colL = wn * 32 + nt * 16 + lr;
      int col = n0 + colL;
#pragma unroll
      for (int i = 0; i < 8; ++i) {
        int lRow = wm * 32 + mt * 16 + lh * 8 + i;
        if (m0 + lRow < M && col < OC) {
          float v = acc[mt][nt][i] + bias[col];
          long oidx = (long)(tNo[lRow] * OC + col) * (OH * OW) + tR[lRow];
          if (outF32) {
            ((float*)out)[oidx] = v;
          } else {
            ((bf16*)out)[oidx] = (bf16)v;
            atomicAdd(&cSum[colL], v);
            atomicAdd(&cSq[colL], v * v);
          }
        }
      }
    }
  }
  if (!outF32) {
    __syncthreads();
    if (tid < TBN && (n0 + tid) < OC) {
      atomicAdd(&statSum[n0 + tid], cSum[tid]);
      atomicAdd(&statSq[n0 + tid], cSq[tid]);
    }
  }
}

// ------------------------------------------------------- transposed conv GEMM

__global__ __launch_bounds__(256) void k_convt_wmma(
    const bf16* __restrict__ in, const bf16* __restrict__ wAll,
    const float* __restrict__ bias, bf16* __restrict__ out,
    float* __restrict__ statSum, float* __restrict__ statSq,
    int Nb, int Cin, int IH, int IW, int OC) {
  __shared__ __align__(16) bf16 As[TBM][TBK];
  __shared__ __align__(32) bf16 Bs[2][TBN][TBK];
  __shared__ float cSum[TBN], cSq[TBN];
  __shared__ int tC[TBM];   // n*Cin
  __shared__ int tY[TBM];   // y (poisoned if invalid)
  __shared__ int tX[TBM];
  __shared__ int tNo[TBM];  // n

  const int tid = threadIdx.x;
  const int p = blockIdx.z;
  const int a = p >> 1, b = p & 1;
  const int K = Cin * 4;
  const bf16* wgt = wAll + (long)p * OC * K;
  const int M  = Nb * IH * IW;
  const int m0 = blockIdx.x * TBM;
  const int n0 = blockIdx.y * TBN;

  if (tid < TBN) { cSum[tid] = 0.0f; cSq[tid] = 0.0f; }
  if (tid < TBM) {
    int m = m0 + tid;
    int n = 0, y = -(1 << 20), x = 0;
    if (m < M) {
      n = m / (IH * IW);
      int r = m - n * (IH * IW);
      y = r / IW;
      x = r - (r / IW) * IW;
    }
    tC[tid] = n * Cin; tY[tid] = y; tX[tid] = x; tNo[tid] = n;
  }
  __syncthreads();

  const int lane = tid & 31;
  const int wave = tid >> 5;
  const int wm = wave & 3, wn = wave >> 2;
  const int lr = lane & 15, lh = lane >> 4;
  const int kiMine = tid & 31;
  const int miBase = tid >> 5;
  const unsigned ldsB0 = (unsigned)(size_t)&Bs[0][0][0];
  const unsigned ldsB1 = (unsigned)(size_t)&Bs[1][0][0];
  const int nk = (K + TBK - 1) / TBK;

  auto loadA = [&](int kt, bf16 (&regs)[16]) {
    int k = kt * TBK + kiMine;
    bool kok = (k < K);
    int ic = k >> 2, t = k & 3, th = t >> 1, tw = t & 1;
    int dy = a ? (th ? 0 : 1) : (th ? -1 : 0);
    int dx = b ? (tw ? 0 : 1) : (tw ? -1 : 0);
#pragma unroll
    for (int i = 0; i < 16; ++i) {
      int mi = i * 8 + miBase;
      int ih = tY[mi] + dy;
      int iw = tX[mi] + dx;
      bf16 v = (bf16)0.0f;
      if (kok && (unsigned)ih < (unsigned)IH && (unsigned)iw < (unsigned)IW)
        v = in[((long)(tC[mi] + ic) * IH + ih) * IW + iw];
      regs[i] = v;
    }
  };

  v8f acc[2][2] = {};
  bf16 aReg[16];
  loadA(0, aReg);
  if (tid == 0)
    tdm_load_b_tile(wgt + (long)n0 * K, K, K, OC - n0, ldsB0);

  for (int kt = 0; kt < nk; ++kt) {
#pragma unroll
    for (int i = 0; i < 16; ++i) As[i * 8 + miBase][kiMine] = aReg[i];
    if (tid == 0) {
      if (kt + 1 < nk) {
        tdm_load_b_tile(wgt + (long)n0 * K + (kt + 1) * TBK, K,
                        K - (kt + 1) * TBK, OC - n0, (kt + 1) & 1 ? ldsB1 : ldsB0);
        __builtin_amdgcn_s_wait_tensorcnt(1);
      } else {
        __builtin_amdgcn_s_wait_tensorcnt(0);
      }
    }
    __syncthreads();
    if (kt + 1 < nk) loadA(kt + 1, aReg);

    v16bf afr[2], bfr[2];
#pragma unroll
    for (int mt = 0; mt < 2; ++mt) {
      int row = wm * 32 + mt * 16 + lr;
      int kh8 = lh * 8;
      v8bf lo = *(const v8bf*)&As[row][2 * kh8];
      v8bf hi = *(const v8bf*)&As[row][2 * kh8 + 8];
#pragma unroll
      for (int j = 0; j < 8; ++j) { afr[mt][j] = lo[j]; afr[mt][8 + j] = hi[j]; }
    }
#pragma unroll
    for (int nt = 0; nt < 2; ++nt) {
      int col = wn * 32 + nt * 16 + lr;
      bfr[nt] = *(const v16bf*)&Bs[kt & 1][col][lh * 16];
    }
#pragma unroll
    for (int mt = 0; mt < 2; ++mt)
#pragma unroll
      for (int nt = 0; nt < 2; ++nt)
        acc[mt][nt] = __builtin_amdgcn_wmma_f32_16x16x32_bf16(
            false, afr[mt], false, bfr[nt], (short)0, acc[mt][nt], false, false);
    __syncthreads();
  }

  const int OH = 2 * IH, OW = 2 * IW;
#pragma unroll
  for (int mt = 0; mt < 2; ++mt) {
#pragma unroll
    for (int nt = 0; nt < 2; ++nt) {
      int colL = wn * 32 + nt * 16 + lr;
      int col = n0 + colL;
#pragma unroll
      for (int i = 0; i < 8; ++i) {
        int lRow = wm * 32 + mt * 16 + lh * 8 + i;
        if (m0 + lRow < M && col < OC) {
          float v = acc[mt][nt][i] + bias[col];
          long oidx = ((long)(tNo[lRow] * OC + col) * OH + (2 * tY[lRow] + a)) * OW +
                      (2 * tX[lRow] + b);
          out[oidx] = (bf16)v;
          atomicAdd(&cSum[colL], v);
          atomicAdd(&cSq[colL], v * v);
        }
      }
    }
  }
  __syncthreads();
  if (tid < TBN && (n0 + tid) < OC) {
    atomicAdd(&statSum[n0 + tid], cSum[tid]);
    atomicAdd(&statSq[n0 + tid], cSq[tid]);
  }
}

// ------------------------------------------------------------- BN + ReLU
// All C and HW in this net are powers of two -> shift/mask, no idiv.
__global__ void k_bn_relu(bf16* __restrict__ x, const float* __restrict__ g,
                          const float* __restrict__ bt,
                          const float* __restrict__ sum,
                          const float* __restrict__ sq,
                          int cMask, int hwShift, long total, float invCnt) {
  long i = (long)blockIdx.x * 256 + threadIdx.x;
  if (i >= total) return;
  int c = (int)((i >> hwShift) & cMask);
  float m = sum[c] * invCnt;
  float v = sq[c] * invCnt - m * m;
  float y = g[c] * ((float)x[i] - m) * rsqrtf(v + 1e-5f) + bt[c];
  x[i] = (bf16)(y > 0.0f ? y : 0.0f);
}

// ---------------------------------------------------------------- VQ

__global__ __launch_bounds__(256) void k_vq(const float* __restrict__ ze,
                                            const float* __restrict__ cbs,
                                            bf16* __restrict__ zq,
                                            float* __restrict__ outIdx,
                                            float* __restrict__ lossAcc) {
  __shared__ float red[256];
  int tIdx = blockIdx.x * 256 + threadIdx.x;
  int n = tIdx & 31;
  int t = tIdx >> 5;
  int bb = t >> 6, hw = t & 63;
  int base = (bb * 256 + n * 8) * 64 + hw;
  float z[8];
#pragma unroll
  for (int j = 0; j < 8; ++j) z[j] = ze[base + j * 64];
  const float* cb = cbs + (long)n * 512 * 8;
  float best = 3.4e38f;
  int bi = 0;
  for (int m = 0; m < 512; ++m) {
    float d = 0.0f;
#pragma unroll
    for (int j = 0; j < 8; ++j) { float df = z[j] - cb[m * 8 + j]; d += df * df; }
    if (d < best) { best = d; bi = m; }
  }
  float err = 0.0f;
#pragma unroll
  for (int j = 0; j < 8; ++j) {
    float q = cb[bi * 8 + j];
    float df = z[j] - q;
    err += df * df;
    zq[base + j * 64] = (bf16)q;  // forward value of straight-through == q
  }
  outIdx[(bb * 32 + n) * 64 + hw] = (float)bi;
  red[threadIdx.x] = err;
  __syncthreads();
  for (int s = 128; s > 0; s >>= 1) {
    if (threadIdx.x < s) red[threadIdx.x] += red[threadIdx.x + s];
    __syncthreads();
  }
  if (threadIdx.x == 0) atomicAdd(lossAcc, red[0]);
}

__global__ void k_loss_finalize(const float* __restrict__ acc, float* __restrict__ out) {
  if (threadIdx.x == 0 && blockIdx.x == 0)
    out[0] = acc[0] * (1.25f / 262144.0f);  // (1+BETA) * mean
}

// ---------------------------------------------------------------- driver

extern "C" void kernel_launch(void* const* d_in, const int* in_sizes, int n_in,
                              void* d_out, int out_size, void* d_ws, size_t ws_size,
                              hipStream_t stream) {
  (void)in_sizes; (void)n_in; (void)out_size; (void)ws_size;
  const float* x = (const float*)d_in[0];
  const float* ew[5]; for (int i = 0; i < 5; ++i) ew[i] = (const float*)d_in[1 + i];
  const float* eb[5]; for (int i = 0; i < 5; ++i) eb[i] = (const float*)d_in[6 + i];
  const float* eg[4]; for (int i = 0; i < 4; ++i) eg[i] = (const float*)d_in[11 + i];
  const float* ebt[4]; for (int i = 0; i < 4; ++i) ebt[i] = (const float*)d_in[15 + i];
  const float* dw[5]; for (int i = 0; i < 5; ++i) dw[i] = (const float*)d_in[19 + i];
  const float* db[5]; for (int i = 0; i < 5; ++i) db[i] = (const float*)d_in[24 + i];
  const float* dg[4]; for (int i = 0; i < 4; ++i) dg[i] = (const float*)d_in[29 + i];
  const float* dbt[4]; for (int i = 0; i < 4; ++i) dbt[i] = (const float*)d_in[33 + i];
  const float* cbs = (const float*)d_in[37];
  float* o = (float*)d_out;

  // ---- workspace carve (~320 MB)
  char* ws = (char*)d_ws;
  size_t off = 0;
  auto carve = [&](size_t bytes) -> char* {
    char* pp = ws + off;
    off = (off + bytes + 255) & ~(size_t)255;
    return pp;
  };
  bf16* actA = (bf16*)carve(33554432ull * 2);
  bf16* actB = (bf16*)carve(67108864ull * 2);
  float* ze  = (float*)carve(262144ull * 4);
  float* sSum = (float*)carve(2048 * 4);
  float* sSq  = sSum + 1024;
  float* lossA = (float*)carve(256);
  bf16* We[5];
  We[0] = (bf16*)carve(12288ull * 2);
  We[1] = (bf16*)carve(2097152ull * 2);
  We[2] = (bf16*)carve(8388608ull * 2);
  We[3] = (bf16*)carve(16777216ull * 2);
  We[4] = (bf16*)carve(262144ull * 2);
  bf16* Wd[5];
  Wd[0] = (bf16*)carve(4194304ull * 2);
  Wd[1] = (bf16*)carve(16777216ull * 2);
  Wd[2] = (bf16*)carve(8388608ull * 2);
  Wd[3] = (bf16*)carve(2097152ull * 2);
  Wd[4] = (bf16*)carve(6912ull * 2);

  auto cvt = [&](const float* s, bf16* d, int n) {
    k_f32_to_bf16<<<(n + 255) / 256, 256, 0, stream>>>(s, d, n);
  };
  cvt(x, actA, 786432);
  cvt(ew[0], We[0], 12288);
  cvt(ew[1], We[1], 2097152);
  cvt(ew[2], We[2], 8388608);
  cvt(ew[3], We[3], 16777216);
  cvt(ew[4], We[4], 262144);
  cvt(dw[4], Wd[4], 6912);
  k_pack_dect<<<(4194304 + 255) / 256, 256, 0, stream>>>(dw[0], Wd[0], 256, 1024);
  k_pack_dect<<<(16777216 + 255) / 256, 256, 0, stream>>>(dw[1], Wd[1], 1024, 1024);
  k_pack_dect<<<(8388608 + 255) / 256, 256, 0, stream>>>(dw[2], Wd[2], 1024, 512);
  k_pack_dect<<<(2097152 + 255) / 256, 256, 0, stream>>>(dw[3], Wd[3], 512, 256);

  auto conv = [&](const bf16* in, const bf16* wgt, const float* bias, void* out,
                  int outF32, int Nb, int Cin, int IH, int IW, int OC,
                  int KH, int KW, int s, int p, const float* g, const float* bt) {
    int OH = (IH + 2 * p - KH) / s + 1;
    int OW = (IW + 2 * p - KW) / s + 1;
    int M = Nb * OH * OW;
    if (!outF32) k_zero_f32<<<8, 256, 0, stream>>>(sSum, 2048);
    dim3 grid((M + TBM - 1) / TBM, (OC + TBN - 1) / TBN);
    k_conv_wmma<<<grid, 256, 0, stream>>>(in, wgt, bias, out, sSum, sSq,
                                          Nb, Cin, IH, IW, OC, OH, OW, KH, KW, s, p, outF32);
    if (!outF32) {
      long total = (long)M * OC;
      k_bn_relu<<<(unsigned)((total + 255) / 256), 256, 0, stream>>>(
          (bf16*)out, g, bt, sSum, sSq, OC - 1, __builtin_ctz(OH * OW), total,
          1.0f / (float)M);
    }
  };
  auto convt = [&](const bf16* in, const bf16* wp, const float* bias, bf16* out,
                   int Nb, int Cin, int IH, int IW, int OC,
                   const float* g, const float* bt) {
    int M = Nb * IH * IW;
    k_zero_f32<<<8, 256, 0, stream>>>(sSum, 2048);
    dim3 grid((M + TBM - 1) / TBM, (OC + TBN - 1) / TBN, 4);
    k_convt_wmma<<<grid, 256, 0, stream>>>(in, wp, bias, out, sSum, sSq,
                                           Nb, Cin, IH, IW, OC);
    long total = (long)Nb * OC * 4 * IH * IW;
    k_bn_relu<<<(unsigned)((total + 255) / 256), 256, 0, stream>>>(
        out, g, bt, sSum, sSq, OC - 1, __builtin_ctz(4 * IH * IW), total,
        1.0f / (float)(4 * M));
  };

  // ---- encoder
  conv(actA, We[0], eb[0], actB, 0, 16, 3, 128, 128, 256, 4, 4, 2, 1, eg[0], ebt[0]);
  conv(actB, We[1], eb[1], actA, 0, 16, 256, 64, 64, 512, 4, 4, 2, 1, eg[1], ebt[1]);
  conv(actA, We[2], eb[2], actB, 0, 16, 512, 32, 32, 1024, 4, 4, 2, 1, eg[2], ebt[2]);
  conv(actB, We[3], eb[3], actA, 0, 16, 1024, 16, 16, 1024, 4, 4, 2, 1, eg[3], ebt[3]);
  conv(actA, We[4], eb[4], ze, 1, 16, 1024, 8, 8, 256, 1, 1, 1, 0, nullptr, nullptr);

  // ---- VQ  (x_r at o[0..786432), loss at o[786432], indices at o[786433..))
  k_zero_f32<<<1, 256, 0, stream>>>(lossA, 1);
  k_vq<<<128, 256, 0, stream>>>(ze, cbs, actB, o + 786433, lossA);
  k_loss_finalize<<<1, 64, 0, stream>>>(lossA, o + 786432);

  // ---- decoder
  convt(actB, Wd[0], db[0], actA, 16, 256, 8, 8, 1024, dg[0], dbt[0]);
  convt(actA, Wd[1], db[1], actB, 16, 1024, 16, 16, 1024, dg[1], dbt[1]);
  convt(actB, Wd[2], db[2], actA, 16, 1024, 32, 32, 512, dg[2], dbt[2]);
  convt(actA, Wd[3], db[3], actB, 16, 512, 64, 64, 256, dg[3], dbt[3]);
  conv(actB, Wd[4], db[4], o, 1, 16, 256, 128, 128, 3, 3, 3, 1, 1, nullptr, nullptr);
}
